// NeuralIntraAttentionModel_56495999811932
// MI455X (gfx1250) — compile-verified, matrix operand order checked by reading, not run on hardware
//
#include <hip/hip_runtime.h>
#include <hip/hip_bf16.h>
#include <math.h>

// ---------------- model dims ----------------
#define V_    32000
#define E_    128
#define H_    160
#define B_    16
#define L_    400
#define T_    64
#define H2_   320
#define G_ENC 640     // 4*H
#define G_DEC 1280    // 4*H2
#define CAT_  960     // 6*H
#define VO_   31997   // V - END
#define VOP_  32000   // padded vocab
#define NT_V  2000    // VOP_/16 vocab tiles
#define KT_V  30      // CAT_/32 k tiles of vocab GEMM
#define PAD_  0
#define NEGV  (-1.0e30f)
#define FRAG  512     // halves per 32x16 / 16x32 fragment

typedef __attribute__((ext_vector_type(16))) _Float16 v16h;
typedef __attribute__((ext_vector_type(8)))  float    v8f;

// ---------------- WMMA helpers (gfx1250, wave32) ----------------
// Fragment-native load: lane's 16 halves contiguous (32B, aligned).
__device__ __forceinline__ v16h ld_frag(const _Float16* F) {
  return *(const v16h*)(F + (threadIdx.x & 31) * 16);
}

// Scatter address for A-fragment layout (single 16-row M tile, element (m,k)):
// lane = m + 16*((k>>3)&1), half = (k&7) + 8*((k>>4)&1), frag = k>>5
__device__ __forceinline__ int a_off(int m, int k) {
  int sub  = k & 31;
  int lane = m + (((sub >> 3) & 1) << 4);
  int half = (sub & 7) + (((sub >> 4) & 1) << 3);
  return (k >> 5) * FRAG + lane * 16 + half;
}

__device__ __forceinline__ v8f wmma16(v16h a, v16h b, v8f c) {
  return __builtin_amdgcn_wmma_f32_16x16x32_f16(false, a, false, b, (short)0, c, false, false);
}

__device__ __forceinline__ float sigm(float x) { return 1.f / (1.f + __expf(-x)); }
// branch-free, saturation-safe tanh
__device__ __forceinline__ float tanh_fast(float x) {
  return 1.f - 2.f / (__expf(2.f * x) + 1.f);
}

// ---------------- block reductions ----------------
__device__ __forceinline__ float block_reduce_max(float v, float* red) {
  int tid = threadIdx.x;
  red[tid] = v; __syncthreads();
  for (int s = blockDim.x >> 1; s > 0; s >>= 1) {
    if (tid < s) red[tid] = fmaxf(red[tid], red[tid + s]);
    __syncthreads();
  }
  float r = red[0]; __syncthreads();
  return r;
}
__device__ __forceinline__ float block_reduce_sum(float v, float* red) {
  int tid = threadIdx.x;
  red[tid] = v; __syncthreads();
  for (int s = blockDim.x >> 1; s > 0; s >>= 1) {
    if (tid < s) red[tid] += red[tid + s];
    __syncthreads();
  }
  float r = red[0]; __syncthreads();
  return r;
}

// ---------------- fragment packers (run once) ----------------
// B = P (K,N) row-major used directly: B[k][n] = P[k][n]. Frags (K/32)x(N/16).
__global__ void k_pack_b(const float* __restrict__ P, _Float16* __restrict__ dst, int K, int N) {
  int idx = blockIdx.x * 256 + threadIdx.x;
  int lane = idx & 31, frag = idx >> 5;
  int KT = K >> 5, NT = N >> 4;
  if (frag >= KT * NT) return;
  int kt = frag / NT, nt = frag - kt * NT;
  int col = lane & 15, kb = (lane >> 4) << 3;
  int k0 = kt * 32, n = nt * 16 + col;
  _Float16* d = dst + (size_t)frag * FRAG + lane * 16;
#pragma unroll
  for (int i = 0; i < 8; ++i) {
    d[i]     = (_Float16)P[(size_t)(k0 + kb + i) * N + n];
    d[i + 8] = (_Float16)P[(size_t)(k0 + kb + 16 + i) * N + n];
  }
}

// B = W^T where W is (N,K) row-major: B[k][n] = W[n][k]. Per-lane contiguous reads.
__global__ void k_pack_bT(const float* __restrict__ W, _Float16* __restrict__ dst, int N, int K) {
  int idx = blockIdx.x * 256 + threadIdx.x;
  int lane = idx & 31, frag = idx >> 5;
  int KT = K >> 5, NT = N >> 4;
  if (frag >= KT * NT) return;
  int kt = frag / NT, nt = frag - kt * NT;
  int col = lane & 15, kb = (lane >> 4) << 3;
  int k0 = kt * 32, n = nt * 16 + col;
  const float* p = W + (size_t)n * K + k0 + kb;
  _Float16* d = dst + (size_t)frag * FRAG + lane * 16;
#pragma unroll
  for (int i = 0; i < 8; ++i) {
    d[i]     = (_Float16)p[i];
    d[i + 8] = (_Float16)p[16 + i];
  }
}

// A = P^T where P is (K,M) row-major: A[m][k] = P[k][m]. Frags (M/16)x(K/32).
__global__ void k_pack_aT(const float* __restrict__ P, _Float16* __restrict__ dst, int M, int K) {
  int idx = blockIdx.x * 256 + threadIdx.x;
  int lane = idx & 31, frag = idx >> 5;
  int MT = M >> 4, KT = K >> 5;
  if (frag >= MT * KT) return;
  int mt = frag / KT, kt = frag - mt * KT;
  int r = lane & 15, kb = (lane >> 4) << 3;
  int m = mt * 16 + r, k0 = kt * 32;
  _Float16* d = dst + (size_t)frag * FRAG + lane * 16;
#pragma unroll
  for (int i = 0; i < 8; ++i) {
    d[i]     = (_Float16)P[(size_t)(k0 + kb + i) * M + m];
    d[i + 8] = (_Float16)P[(size_t)(k0 + kb + 16 + i) * M + m];
  }
}

// B = emb[3:]^T : B[k=e][n=v] = emb[(3+v)*E + e]; frags 4 x 2000, zero for v >= VO_.
__global__ void k_pack_embT(const float* __restrict__ emb, _Float16* __restrict__ dst) {
  int idx = blockIdx.x * 256 + threadIdx.x;
  int lane = idx & 31, frag = idx >> 5;
  if (frag >= 4 * NT_V) return;
  int kt = frag / NT_V, nt = frag - kt * NT_V;
  int col = lane & 15, kb = (lane >> 4) << 3;
  int v = nt * 16 + col, k0 = kt * 32;
  _Float16* d = dst + (size_t)frag * FRAG + lane * 16;
  if (v < VO_) {
    const float* p = emb + (size_t)(3 + v) * E_ + k0 + kb;
#pragma unroll
    for (int i = 0; i < 8; ++i) { d[i] = (_Float16)p[i]; d[i + 8] = (_Float16)p[16 + i]; }
  } else {
#pragma unroll
    for (int i = 0; i < 16; ++i) d[i] = (_Float16)0.f;
  }
}

__global__ void k_embed_enc(const int* __restrict__ ids, const float* __restrict__ emb,
                            _Float16* __restrict__ x) {
  int idx = blockIdx.x * 256 + threadIdx.x;
  if (idx >= B_ * L_ * E_) return;
  int tok = idx / E_, e = idx - tok * E_;
  int id = ids[tok];
  if (id >= V_ || id < 0) id = 1;  // UNK
  x[idx] = (_Float16)emb[id * E_ + e];
}

// Build vocab-GEMM B fragments directly:
// C' = vocab_proj^T (960x128) @ emb[3:]^T (128x32000); tanh; a wave's cat tiles
// {2p,2p+1} for one vocab tile are exactly k-fragment p -> one 32B store per lane.
__global__ void k_outprojT(const _Float16* __restrict__ vprojA,  // A frags 60x4
                           const _Float16* __restrict__ embB,    // B frags 4x2000
                           _Float16* __restrict__ outTF) {       // B frags 30x2000
  int nt = blockIdx.x;  // vocab tile
  int wave = threadIdx.x >> 5, lane = threadIdx.x & 31;
  for (int p = wave; p < KT_V; p += 8) {
    v8f acc0 = {}, acc1 = {};
#pragma unroll
    for (int k = 0; k < 4; ++k) {
      v16h b  = ld_frag(embB + ((size_t)k * NT_V + nt) * FRAG);
      v16h a0 = ld_frag(vprojA + ((size_t)(2 * p) * 4 + k) * FRAG);
      v16h a1 = ld_frag(vprojA + ((size_t)(2 * p + 1) * 4 + k) * FRAG);
      acc0 = wmma16(a0, b, acc0);
      acc1 = wmma16(a1, b, acc1);
    }
    v16h o;
#pragma unroll
    for (int r = 0; r < 8; ++r) {
      o[r]     = (_Float16)tanh_fast(acc0[r]);
      o[r + 8] = (_Float16)tanh_fast(acc1[r]);
    }
    *(v16h*)(outTF + ((size_t)p * NT_V + nt) * FRAG + lane * 16) = o;
  }
}

__global__ void k_init(float* cum, float* cdec, float* nll, int* cur) {
  int i = blockIdx.x * 256 + threadIdx.x;
  if (i < B_ * L_)  cum[i]  = 0.f;
  if (i < B_ * H2_) cdec[i] = 0.f;
  if (i < B_) { nll[i] = 0.f; cur[i] = 2; }  // START
}

// ---------------- persistent bidirectional LSTM encoder ----------------
// grid = 2 (dir), block = 256 (8 waves). State in LDS in A-frag layout.
__global__ void k_encoder(const _Float16* __restrict__ x_enc,
                          const _Float16* __restrict__ WihF, const _Float16* __restrict__ WhhF,
                          const float* __restrict__ b_f,
                          const _Float16* __restrict__ WihB, const _Float16* __restrict__ WhhB,
                          const float* __restrict__ b_b,
                          float* __restrict__ enc_h, float* __restrict__ h_dec,
                          _Float16* __restrict__ h_dec16) {
  const int dir = blockIdx.x;
  const _Float16* Wih = dir ? WihB : WihF;   // frags 4 x 40
  const _Float16* Whh = dir ? WhhB : WhhF;   // frags 5 x 40
  const float* bias   = dir ? b_b  : b_f;

  __shared__ __align__(32) _Float16 sh_x[4 * FRAG];  // A frags: x_t (16x128)
  __shared__ __align__(32) _Float16 sh_h[5 * FRAG];  // A frags: h   (16x160)
  __shared__ float sh_c[B_ * H_];

  int tid = threadIdx.x, wave = tid >> 5, lane = tid & 31;
  for (int i = tid; i < 5 * FRAG; i += 256) sh_h[i] = (_Float16)0.f;
  for (int i = tid; i < B_ * H_; i += 256) sh_c[i] = 0.f;
  __syncthreads();

  for (int s = 0; s < L_; ++s) {
    int l = dir ? (L_ - 1 - s) : s;
    for (int i = tid; i < B_ * E_; i += 256) {
      int b = i >> 7, e = i & 127;
      sh_x[a_off(b, e)] = x_enc[(b * L_ + l) * E_ + e];
    }
    __syncthreads();

    v8f acc[2][4];
#pragma unroll
    for (int it = 0; it < 2; ++it) {
      int j = wave + it * 8;               // hidden tile (10)
      if (j < H_ / 16) {
#pragma unroll
        for (int g = 0; g < 4; ++g) {
          v8f a = {};
          int nt = g * (H_ / 16) + j;      // of 40
#pragma unroll
          for (int k = 0; k < E_ / 32; ++k)
            a = wmma16(ld_frag(sh_x + k * FRAG), ld_frag(Wih + ((size_t)k * 40 + nt) * FRAG), a);
#pragma unroll
          for (int k = 0; k < H_ / 32; ++k)
            a = wmma16(ld_frag(sh_h + k * FRAG), ld_frag(Whh + ((size_t)k * 40 + nt) * FRAG), a);
          acc[it][g] = a;
        }
      }
    }
    __syncthreads();   // all reads of sh_h done before writes

#pragma unroll
    for (int it = 0; it < 2; ++it) {
      int j = wave + it * 8;
      if (j < H_ / 16) {
        int col = lane & 15, rb = (lane >> 4) << 3;
        int hu = j * 16 + col;
        float bi = bias[hu], bfv = bias[H_ + hu], bg = bias[2 * H_ + hu], bo = bias[3 * H_ + hu];
#pragma unroll
        for (int r = 0; r < 8; ++r) {
          int b = rb + r;
          float ig = sigm(acc[it][0][r] + bi);
          float fg = sigm(acc[it][1][r] + bfv);
          float gg = tanh_fast(acc[it][2][r] + bg);
          float og = sigm(acc[it][3][r] + bo);
          float c = fg * sh_c[b * H_ + hu] + ig * gg;
          float h = og * tanh_fast(c);
          sh_c[b * H_ + hu] = c;
          sh_h[a_off(b, hu)] = (_Float16)h;
          enc_h[(b * L_ + l) * H2_ + dir * H_ + hu] = h;
          if (s == L_ - 1) {
            h_dec[b * H2_ + dir * H_ + hu] = h;
            h_dec16[a_off(b, dir * H_ + hu)] = (_Float16)h;   // A-frag layout (10 frags)
          }
        }
      }
    }
    __syncthreads();
  }
}

// ---------------- decoder step kernels ----------------
__global__ void k_dec_embed(const int* __restrict__ cur, const float* __restrict__ emb,
                            _Float16* __restrict__ x_frag) {
  int b = blockIdx.x, e = threadIdx.x;
  x_frag[a_off(b, e)] = (_Float16)emb[cur[b] * E_ + e];
}

// one block, 256 threads. 20 hidden tiles x 4 gates. All operands frag-native.
__global__ void k_dec_lstm(int t, const _Float16* __restrict__ x_frag, _Float16* h16,
                           const _Float16* __restrict__ Wih, const _Float16* __restrict__ Whh,
                           const float* __restrict__ bias, float* __restrict__ hf,
                           float* __restrict__ cf, float* __restrict__ buf,
                           _Float16* __restrict__ cat) {
  int wave = threadIdx.x >> 5, lane = threadIdx.x & 31;
  v8f acc[3][4];
#pragma unroll
  for (int it = 0; it < 3; ++it) {
    int j = wave + it * 8;
    if (j < H2_ / 16) {
#pragma unroll
      for (int g = 0; g < 4; ++g) {
        v8f a = {};
        int nt = g * (H2_ / 16) + j;       // of 80
#pragma unroll
        for (int k = 0; k < E_ / 32; ++k)
          a = wmma16(ld_frag(x_frag + k * FRAG), ld_frag(Wih + ((size_t)k * 80 + nt) * FRAG), a);
#pragma unroll
        for (int k = 0; k < H2_ / 32; ++k)
          a = wmma16(ld_frag(h16 + k * FRAG), ld_frag(Whh + ((size_t)k * 80 + nt) * FRAG), a);
        acc[it][g] = a;
      }
    }
  }
  __syncthreads();   // all reads of h16 complete before updates

#pragma unroll
  for (int it = 0; it < 3; ++it) {
    int j = wave + it * 8;
    if (j < H2_ / 16) {
      int col = lane & 15, rb = (lane >> 4) << 3;
      int hu = j * 16 + col;
      float bi = bias[hu], bfv = bias[H2_ + hu], bg = bias[2 * H2_ + hu], bo = bias[3 * H2_ + hu];
#pragma unroll
      for (int r = 0; r < 8; ++r) {
        int b = rb + r;
        float ig = sigm(acc[it][0][r] + bi);
        float fg = sigm(acc[it][1][r] + bfv);
        float gg = tanh_fast(acc[it][2][r] + bg);
        float og = sigm(acc[it][3][r] + bo);
        float c = fg * cf[b * H2_ + hu] + ig * gg;
        float h = og * tanh_fast(c);
        cf[b * H2_ + hu] = c;
        hf[b * H2_ + hu] = h;
        h16[a_off(b, hu)] = (_Float16)h;
        buf[(b * T_ + t) * H2_ + hu] = h;
        cat[a_off(b, hu)] = (_Float16)h;   // cat k-range [0,H2)
      }
    }
  }
}

// C(16xN) = A(16xK) @ B(KxN); A frag-native (K/32 frags), B frags (K/32)x(N/16).
__global__ void k_gemm16(const _Float16* __restrict__ A, const _Float16* __restrict__ Bm,
                         float* __restrict__ C, int K, int N) {
  int wave = threadIdx.x >> 5, lane = threadIdx.x & 31;
  int NT = N >> 4;
  for (int nt = wave; nt < NT; nt += 8) {
    v8f acc = {};
    for (int k = 0; k < K / 32; ++k)
      acc = wmma16(ld_frag(A + k * FRAG), ld_frag(Bm + ((size_t)k * NT + nt) * FRAG), acc);
    int col = lane & 15, rb = (lane >> 4) << 3;
#pragma unroll
    for (int r = 0; r < 8; ++r) C[(rb + r) * N + nt * 16 + col] = acc[r];
  }
}

// scores[b][l] = q[b] . enc_h[b][l]; one wave per (b,l)
__global__ void k_attn_scores(const float* __restrict__ q, const float* __restrict__ enc_h,
                              float* __restrict__ scores) {
  int pair = blockIdx.x * 8 + (threadIdx.x >> 5);
  int lane = threadIdx.x & 31;
  int b = pair / L_, l = pair - b * L_;
  float p = 0.f;
  for (int d = lane; d < H2_; d += 32) p += q[b * H2_ + d] * enc_h[(b * L_ + l) * H2_ + d];
  for (int off = 16; off; off >>= 1) p += __shfl_down(p, off, 32);
  if (lane == 0) scores[b * L_ + l] = p;
}

// temporal attention + context + pointer prob + cum update; one block per b
__global__ void k_enc_softmax(int t, const float* __restrict__ scores, float* __restrict__ cum,
                              const int* __restrict__ input_ids, const int* __restrict__ target_ids,
                              const float* __restrict__ enc_h, float* __restrict__ enc_ctx,
                              float* __restrict__ ptr_p, _Float16* __restrict__ cat) {
  int b = blockIdx.x, tid = threadIdx.x;
  __shared__ float s_att[L_];
  __shared__ float red[256];
  int tgt = target_ids[b * T_ + t];
  float lmax = -3.4e38f;
  for (int l = tid; l < L_; l += 256) {
    float sc = scores[b * L_ + l];
    float cv = cum[b * L_ + l];
    float tv = (t == 0) ? sc : (__expf(sc) / cv);
    if (input_ids[b * L_ + l] == PAD_) tv = NEGV;
    s_att[l] = tv;
    cum[b * L_ + l] = cv + sc;
    lmax = fmaxf(lmax, tv);
  }
  __syncthreads();
  float mx = block_reduce_max(lmax, red);
  float lsum = 0.f, lp = 0.f;
  for (int l = tid; l < L_; l += 256) {
    float e = __expf(s_att[l] - mx);
    s_att[l] = e;
    lsum += e;
    if (input_ids[b * L_ + l] == tgt) lp += e;
  }
  float sum = block_reduce_sum(lsum, red);
  float inv = 1.f / sum;
  float pp = block_reduce_sum(lp, red);
  if (tid == 0) ptr_p[b] = pp * inv;
  for (int d = tid; d < H2_; d += 256) {
    float a = 0.f;
    for (int l = 0; l < L_; ++l) a += s_att[l] * enc_h[(b * L_ + l) * H2_ + d];
    a *= inv;
    enc_ctx[b * H2_ + d] = a;
    cat[a_off(b, H2_ + d)] = (_Float16)a;
  }
}

// intra-decoder attention; one block per b
__global__ void k_dec_attn(int t, const float* __restrict__ q2, const float* __restrict__ buf,
                           float* __restrict__ dec_ctx, _Float16* __restrict__ cat) {
  int b = blockIdx.x, tid = threadIdx.x, wave = tid >> 5, lane = tid & 31;
  __shared__ float s2[T_];
  __shared__ float red[256];
  if (t == 0) {
    for (int d = tid; d < H2_; d += 256) {
      dec_ctx[b * H2_ + d] = 0.f;
      cat[a_off(b, 2 * H2_ + d)] = (_Float16)0.f;
    }
    return;
  }
  for (int tt = wave; tt < t; tt += 8) {
    float p = 0.f;
    for (int d = lane; d < H2_; d += 32) p += q2[b * H2_ + d] * buf[(b * T_ + tt) * H2_ + d];
    for (int off = 16; off; off >>= 1) p += __shfl_down(p, off, 32);
    if (lane == 0) s2[tt] = p;
  }
  __syncthreads();
  float lm = -3.4e38f;
  for (int tt = tid; tt < t; tt += 256) lm = fmaxf(lm, s2[tt]);
  float mx = block_reduce_max(lm, red);
  float ls = 0.f;
  for (int tt = tid; tt < t; tt += 256) { float e = __expf(s2[tt] - mx); s2[tt] = e; ls += e; }
  float sum = block_reduce_sum(ls, red);
  float inv = 1.f / sum;
  for (int d = tid; d < H2_; d += 256) {
    float a = 0.f;
    for (int tt = 0; tt < t; ++tt) a += s2[tt] * buf[(b * T_ + tt) * H2_ + d];
    a *= inv;
    dec_ctx[b * H2_ + d] = a;
    cat[a_off(b, 2 * H2_ + d)] = (_Float16)a;
  }
}

__global__ void k_switch(const float* __restrict__ hf, const float* __restrict__ enc_ctx,
                         const float* __restrict__ dec_ctx, const float* __restrict__ sw,
                         const float* __restrict__ sb, float* __restrict__ p_gen) {
  int b = blockIdx.x, tid = threadIdx.x;
  __shared__ float red[256];
  float lv = 0.f;
  for (int d = tid; d < CAT_; d += 256) {
    float x = (d < H2_) ? hf[b * H2_ + d]
            : (d < 2 * H2_) ? enc_ctx[b * H2_ + d - H2_]
                            : dec_ctx[b * H2_ + d - 2 * H2_];
    lv += x * sw[d];
  }
  float s = block_reduce_sum(lv, red);
  if (tid == 0) p_gen[b] = sigm(s + sb[0]);
}

// vocab logits: cat(16x960, A frags) @ out_projT frags (30x2000) + bias.
// Each wave computes 2 adjacent vocab tiles, sharing A loads.
__global__ void k_vocab(const _Float16* __restrict__ cat, const _Float16* __restrict__ outTF,
                        const float* __restrict__ out_bias, float* __restrict__ logits) {
  int wv = blockIdx.x * 8 + (threadIdx.x >> 5);
  int lane = threadIdx.x & 31;
  int t0 = wv * 2;
  if (t0 >= NT_V) return;
  v8f acc0 = {}, acc1 = {};
#pragma unroll
  for (int k = 0; k < KT_V; ++k) {
    if (k + 1 < KT_V)
      __builtin_prefetch(outTF + ((size_t)(k + 1) * NT_V + t0) * FRAG + lane * 16, 0, 0);
    v16h af = ld_frag(cat + k * FRAG);
    acc0 = wmma16(af, ld_frag(outTF + ((size_t)k * NT_V + t0) * FRAG), acc0);
    acc1 = wmma16(af, ld_frag(outTF + ((size_t)k * NT_V + t0 + 1) * FRAG), acc1);
  }
  int col = lane & 15, rb = (lane >> 4) << 3;
#pragma unroll
  for (int s = 0; s < 2; ++s) {
    int vi = (t0 + s) * 16 + col;
    float bias = (vi < VO_) ? out_bias[vi] : 0.f;
#pragma unroll
    for (int r = 0; r < 8; ++r) {
      float aval = s ? acc1[r] : acc0[r];
      float val = (vi < VO_) ? (aval + bias) : NEGV;
      logits[(rb + r) * VOP_ + vi] = val;
    }
  }
}

// softmax normalizer over vocab + target prob; one block per b
__global__ void k_vred(int t, const float* __restrict__ logits, const int* __restrict__ target_ids,
                       float* __restrict__ gen_p) {
  int b = blockIdx.x, tid = threadIdx.x;
  __shared__ float red[256];
  float lm = -3.4e38f;
  for (int v = tid; v < VOP_; v += 256) lm = fmaxf(lm, logits[b * VOP_ + v]);
  float mx = block_reduce_max(lm, red);
  float ls = 0.f;
  for (int v = tid; v < VOP_; v += 256) ls += __expf(logits[b * VOP_ + v] - mx);
  float sum = block_reduce_sum(ls, red);
  if (tid == 0) {
    int tgt = target_ids[b * T_ + t];
    int gi = tgt - 3;  // END offset
    float gp = 0.f;
    if (gi >= 0 && gi < VO_) gp = __expf(logits[b * VOP_ + gi] - mx) / sum;
    gen_p[b] = gp;
  }
}

__global__ void k_nll(int t, const float* __restrict__ p_gen, const float* __restrict__ gen_p,
                      const float* __restrict__ ptr_p, const int* __restrict__ target_ids,
                      const int* __restrict__ input_lengths, float* __restrict__ nll,
                      int* __restrict__ cur) {
  int b = threadIdx.x;
  if (b < B_) {
    float pg = p_gen[b];
    float p = pg * gen_p[b] + (1.f - pg) * ptr_p[b];
    if (input_lengths[b] > t) nll[b] += -__logf(p + 1e-9f);
    int tgt = target_ids[b * T_ + t];
    cur[b] = (tgt >= V_) ? 1 : tgt;
  }
}

__global__ void k_final(const float* __restrict__ nll, float* __restrict__ out) {
  float v = (threadIdx.x < B_) ? nll[threadIdx.x] : 0.f;
  for (int off = 16; off; off >>= 1) v += __shfl_down(v, off, 32);
  if (threadIdx.x == 0) out[0] = v;
}

// ---------------- host launch ----------------
extern "C" void kernel_launch(void* const* d_in, const int* in_sizes, int n_in,
                              void* d_out, int out_size, void* d_ws, size_t ws_size,
                              hipStream_t stream) {
  const int*   input_ids     = (const int*)d_in[0];
  const int*   target_ids    = (const int*)d_in[1];
  const int*   input_lengths = (const int*)d_in[2];
  const float* emb           = (const float*)d_in[3];
  const float* enc_Wih_f     = (const float*)d_in[4];
  const float* enc_Whh_f     = (const float*)d_in[5];
  const float* enc_b_f       = (const float*)d_in[6];
  const float* enc_Wih_b     = (const float*)d_in[7];
  const float* enc_Whh_b     = (const float*)d_in[8];
  const float* enc_b_b       = (const float*)d_in[9];
  const float* dec_Wih       = (const float*)d_in[10];
  const float* dec_Whh       = (const float*)d_in[11];
  const float* dec_b         = (const float*)d_in[12];
  const float* enc_attn_proj = (const float*)d_in[13];
  const float* dec_attn_proj = (const float*)d_in[14];
  const float* vocab_proj    = (const float*)d_in[15];
  const float* switch_w      = (const float*)d_in[16];
  const float* switch_b      = (const float*)d_in[17];
  const float* out_bias      = (const float*)d_in[18];
  float* out = (float*)d_out;

  char* w = (char*)d_ws;
  size_t off = 0;
  auto take = [&](size_t bytes) -> void* {
    void* p = w + off;
    off = (off + bytes + 255) & ~(size_t)255;
    return p;
  };

  // fragment-packed weights (sizes = K*N halves)
  _Float16* wEncWihF = (_Float16*)take((size_t)E_  * G_ENC * 2);
  _Float16* wEncWhhF = (_Float16*)take((size_t)H_  * G_ENC * 2);
  _Float16* wEncWihB = (_Float16*)take((size_t)E_  * G_ENC * 2);
  _Float16* wEncWhhB = (_Float16*)take((size_t)H_  * G_ENC * 2);
  _Float16* wDecWih  = (_Float16*)take((size_t)E_  * G_DEC * 2);
  _Float16* wDecWhh  = (_Float16*)take((size_t)H2_ * G_DEC * 2);
  _Float16* wEncP    = (_Float16*)take((size_t)H2_ * H2_ * 2);
  _Float16* wDecP    = (_Float16*)take((size_t)H2_ * H2_ * 2);
  _Float16* vprojA   = (_Float16*)take((size_t)CAT_ * E_ * 2);          // A frags 60x4
  _Float16* embB     = (_Float16*)take((size_t)4 * NT_V * FRAG * 2);    // B frags 4x2000
  _Float16* outTF    = (_Float16*)take((size_t)KT_V * NT_V * FRAG * 2); // 61.4 MB, L2 resident
  _Float16* x_enc    = (_Float16*)take((size_t)B_ * L_ * E_ * 2);
  float*    enc_h    = (float*)take((size_t)B_ * L_ * H2_ * 4);
  float*    h_dec    = (float*)take((size_t)B_ * H2_ * 4);
  float*    c_dec    = (float*)take((size_t)B_ * H2_ * 4);
  _Float16* h16      = (_Float16*)take((size_t)B_ * H2_ * 2);   // A frags (10)
  float*    cum      = (float*)take((size_t)B_ * L_ * 4);
  float*    buf      = (float*)take((size_t)B_ * T_ * H2_ * 4);
  float*    scores   = (float*)take((size_t)B_ * L_ * 4);
  float*    q        = (float*)take((size_t)B_ * H2_ * 4);
  float*    q2       = (float*)take((size_t)B_ * H2_ * 4);
  float*    enc_ctx  = (float*)take((size_t)B_ * H2_ * 4);
  float*    dec_ctx  = (float*)take((size_t)B_ * H2_ * 4);
  float*    ptr_p    = (float*)take((size_t)B_ * 4);
  float*    p_gen    = (float*)take((size_t)B_ * 4);
  float*    gen_p    = (float*)take((size_t)B_ * 4);
  float*    nll      = (float*)take((size_t)B_ * 4);
  int*      cur      = (int*)take((size_t)B_ * 4);
  _Float16* cat16    = (_Float16*)take((size_t)B_ * CAT_ * 2);  // A frags (30)
  _Float16* x_dec    = (_Float16*)take((size_t)B_ * E_ * 2);    // A frags (4)
  float*    logits   = (float*)take((size_t)B_ * VOP_ * 4);
  (void)ws_size; (void)n_in; (void)in_sizes; (void)out_size;

  // ---- pack weights into WMMA fragment layout (once) ----
  k_pack_bT<<<(4 * 40 * 32 + 255) / 256, 256, 0, stream>>>(enc_Wih_f, wEncWihF, G_ENC, E_);
  k_pack_bT<<<(5 * 40 * 32 + 255) / 256, 256, 0, stream>>>(enc_Whh_f, wEncWhhF, G_ENC, H_);
  k_pack_bT<<<(4 * 40 * 32 + 255) / 256, 256, 0, stream>>>(enc_Wih_b, wEncWihB, G_ENC, E_);
  k_pack_bT<<<(5 * 40 * 32 + 255) / 256, 256, 0, stream>>>(enc_Whh_b, wEncWhhB, G_ENC, H_);
  k_pack_bT<<<(4 * 80 * 32 + 255) / 256, 256, 0, stream>>>(dec_Wih, wDecWih, G_DEC, E_);
  k_pack_bT<<<(10 * 80 * 32 + 255) / 256, 256, 0, stream>>>(dec_Whh, wDecWhh, G_DEC, H2_);
  k_pack_b<<<(10 * 20 * 32 + 255) / 256, 256, 0, stream>>>(enc_attn_proj, wEncP, H2_, H2_);
  k_pack_b<<<(10 * 20 * 32 + 255) / 256, 256, 0, stream>>>(dec_attn_proj, wDecP, H2_, H2_);
  k_pack_aT<<<(60 * 4 * 32 + 255) / 256, 256, 0, stream>>>(vocab_proj, vprojA, CAT_, E_);
  k_pack_embT<<<(4 * NT_V * 32 + 255) / 256, 256, 0, stream>>>(emb, embB);
  k_embed_enc<<<(B_ * L_ * E_ + 255) / 256, 256, 0, stream>>>(input_ids, emb, x_enc);
  k_outprojT<<<NT_V, 256, 0, stream>>>(vprojA, embB, outTF);
  k_init<<<(B_ * L_ + 255) / 256, 256, 0, stream>>>(cum, c_dec, nll, cur);

  // ---- encoder (persistent, 1 WG per direction) ----
  k_encoder<<<2, 256, 0, stream>>>(x_enc, wEncWihF, wEncWhhF, enc_b_f,
                                   wEncWihB, wEncWhhB, enc_b_b, enc_h, h_dec, h16);

  // ---- decoder steps ----
  for (int t = 0; t < T_; ++t) {
    k_dec_embed<<<B_, E_, 0, stream>>>(cur, emb, x_dec);
    k_dec_lstm<<<1, 256, 0, stream>>>(t, x_dec, h16, wDecWih, wDecWhh, dec_b,
                                      h_dec, c_dec, buf, cat16);
    k_gemm16<<<1, 256, 0, stream>>>(h16, wEncP, q, H2_, H2_);
    k_attn_scores<<<(B_ * L_) / 8, 256, 0, stream>>>(q, enc_h, scores);
    k_enc_softmax<<<B_, 256, 0, stream>>>(t, scores, cum, input_ids, target_ids,
                                          enc_h, enc_ctx, ptr_p, cat16);
    k_gemm16<<<1, 256, 0, stream>>>(h16, wDecP, q2, H2_, H2_);
    k_dec_attn<<<B_, 256, 0, stream>>>(t, q2, buf, dec_ctx, cat16);
    k_switch<<<B_, 256, 0, stream>>>(h_dec, enc_ctx, dec_ctx, switch_w, switch_b, p_gen);
    k_vocab<<<NT_V / 16, 256, 0, stream>>>(cat16, outTF, out_bias, logits);
    k_vred<<<B_, 256, 0, stream>>>(t, logits, target_ids, gen_p);
    k_nll<<<1, 32, 0, stream>>>(t, p_gen, gen_p, ptr_p, target_ids, input_lengths, nll, cur);
  }

  k_final<<<1, 32, 0, stream>>>(nll, out);
}
